// MultiheadAttention_90838558310744
// MI455X (gfx1250) — matrix operand order repaired and checked by reading, not compiled, not run on hardware
//
#include <hip/hip_runtime.h>
#include <hip/hip_bf16.h>

// ---------------------------------------------------------------------------
// Transformer-XL relative multi-head attention for MI455X (gfx1250, wave32).
// All GEMMs on v_wmma_f32_16x16x32_bf16 (f32 accumulate). Operand staging uses
// GLOBAL_LOAD_ASYNC_TO_LDS_B128 (ASYNCcnt DMA, no VGPR round trip), double-
// buffered LDS, one s_wait_asynccnt + barrier per K-stage. Every GEMM is NT
// (B stored N-major, K-contiguous), which the weight/v transposes arrange.
// Pipeline:
//   1) x,pos: f32->bf16 ; Wq/Wk/Wv/Wr/merge_W: f32->bf16 TRANSPOSED
//   2) projections (NT): q (f32,*scale), k (bf16 [B,T,H,D]),
//      v (bf16, C-transposed -> vT[H*D][B*T]), r (bf16 [P,H,D])
//   3) q + r_w_bias -> qw (bf16), q + r_r_bias -> qr (bf16)
//   4) content = qw.k^T (f32), relative = qr.r^T (bf16)   [batched NT]
//   5) softmax(content + rel_shift(relative)); bf16 weights in place
//   6) attended = weights.v (NT via vT, bf16 [B,T,H,D])
//   7) out = attended @ merge_W + merge_b (NT via W^T, f32 to d_out)
// ---------------------------------------------------------------------------

typedef __attribute__((ext_vector_type(16))) __bf16 v16bf;
typedef __attribute__((ext_vector_type(8)))  __bf16 v8bf;
typedef __attribute__((ext_vector_type(4)))  __bf16 v4bf;
typedef __attribute__((ext_vector_type(8)))  float  v8f;
typedef __attribute__((ext_vector_type(4)))  float  v4f;

#define CAT16(lo, hi) __builtin_shufflevector(lo, hi, 0,1,2,3,4,5,6,7,8,9,10,11,12,13,14,15)

static constexpr int Bb = 4;
static constexpr int Tt = 1024;
static constexpr int Ee = 1024;
static constexpr int Hh = 16;
static constexpr int Dd = 64;
static constexpr long BT = (long)Bb * Tt;   // 4096
static constexpr long BH = (long)Bb * Hh;   // 64

// --------------------------- elementwise kernels ---------------------------

__global__ __launch_bounds__(256) void cvt_f32_bf16(const float* __restrict__ s,
                                                    __bf16* __restrict__ d, long n4) {
  long i = (long)blockIdx.x * 256 + threadIdx.x;
  long stride = (long)gridDim.x * 256;
  for (; i < n4; i += stride) {
    v4f v = ((const v4f*)s)[i];
    v4bf o;
#pragma unroll
    for (int j = 0; j < 4; ++j) o[j] = (__bf16)v[j];
    ((v4bf*)d)[i] = o;
  }
}

// Transposing convert: d[c][r] = (bf16) s[r][c]; R, C multiples of 32.
__global__ __launch_bounds__(256) void cvt_t_f32_bf16(const float* __restrict__ s,
                                                      __bf16* __restrict__ d,
                                                      int R, int C) {
  __shared__ float tile[32][33];
  const int r0 = blockIdx.y * 32, c0 = blockIdx.x * 32;
  const int tx = threadIdx.x, ty = threadIdx.y;          // 32 x 8
#pragma unroll
  for (int j = 0; j < 32; j += 8)
    tile[ty + j][tx] = s[(long)(r0 + ty + j) * C + (c0 + tx)];
  __syncthreads();
#pragma unroll
  for (int j = 0; j < 32; j += 8)
    d[(long)(c0 + ty + j) * R + (r0 + tx)] = (__bf16)tile[tx][ty + j];
}

__global__ __launch_bounds__(256) void qbias_kernel(const float* __restrict__ q,
                                                    const float* __restrict__ bw,
                                                    const float* __restrict__ br,
                                                    __bf16* __restrict__ qw,
                                                    __bf16* __restrict__ qr, long n4) {
  long i = (long)blockIdx.x * 256 + threadIdx.x;
  long stride = (long)gridDim.x * 256;
  for (; i < n4; i += stride) {
    int c4 = (int)(i & (Ee / 4 - 1));   // vector column group within E
    v4f v = ((const v4f*)q)[i];
    v4f w = ((const v4f*)bw)[c4];
    v4f r = ((const v4f*)br)[c4];
    v4bf ow, orr;
#pragma unroll
    for (int j = 0; j < 4; ++j) {
      ow[j]  = (__bf16)(v[j] + w[j]);
      orr[j] = (__bf16)(v[j] + r[j]);
    }
    ((v4bf*)qw)[i] = ow;
    ((v4bf*)qr)[i] = orr;
  }
}

// ----------------------------- NT WMMA GEMM --------------------------------
// C[g] = alpha * A[g] . B[g]^T (+bias[col]).  A: M x K row-major (lda).
// B: N-major, K-contiguous: element (k,n) at B[n*ldb + k].
// Batch offset for index g: (g/batchInner)*strideOuter + (g%batchInner)*strideInner.
// outBf16: bf16 vs f32 C. storeCT: store C transposed (C[n*ldc + m], bf16 only).
// M%BM==0, N%BN==0, K%64==0 for all calls; all tiles 16B aligned.

struct GemmArgs {
  const __bf16* A;
  const __bf16* B;
  void* C;
  const float* bias;     // nullable, f32, indexed by global column
  long lda, ldb, ldc;
  long sAo, sAi, sBo, sBi, sCo, sCi;
  int batchInner;
  int K;
  float alpha;
  int outBf16;
  int storeCT;
};

template <int BM, int BN, int WR, int WC>
__global__ __launch_bounds__(256) void gemm_bf16_nt_kernel(GemmArgs p) {
  constexpr int KT = 64;             // K per stage = two WMMA K-slices
  constexpr int LDSP = KT + 8;       // padded row stride: 72 bf16 = 144 B
  constexpr int WM = BM / WR;
  constexpr int WN = BN / WC;
  constexpr int TI = WM / 16;
  constexpr int TJ = WN / 16;
  constexpr int AV = (BM * KT) / (8 * 256);   // 16 B DMA ops / thread for A
  constexpr int BV = (BN * KT) / (8 * 256);

  __shared__ __align__(16) __bf16 ldsA[2][BM * LDSP];   // [m][k]
  __shared__ __align__(16) __bf16 ldsB[2][BN * LDSP];   // [n][k]

  const int tid  = threadIdx.x;
  const int wave = tid >> 5;
  const int lane = tid & 31;
  const int kh   = lane >> 4;        // lane half per WMMA operand layout
  const int l16  = lane & 15;
  const int wr   = wave / WC;
  const int wc   = wave % WC;

  const int g = blockIdx.z;
  const long offA = (long)(g / p.batchInner) * p.sAo + (long)(g % p.batchInner) * p.sAi;
  const long offB = (long)(g / p.batchInner) * p.sBo + (long)(g % p.batchInner) * p.sBi;
  const long offC = (long)(g / p.batchInner) * p.sCo + (long)(g % p.batchInner) * p.sCi;
  const __bf16* Ag = p.A + offA;
  const __bf16* Bg = p.B + offB;

  const int m0 = blockIdx.y * BM;
  const int n0 = blockIdx.x * BN;

  v8f acc[TI][TJ] = {};

  // ---- async DMA stage: global -> LDS, no VGPR round trip (ASYNCcnt) ----
  auto stage = [&](int buf, int k0) {
#pragma unroll
    for (int it = 0; it < AV; ++it) {
      int s = tid + 256 * it;
      int row = s >> 3;              // KT/8 = 8 16-byte copies per row
      int kq = (s & 7) << 3;
      const __bf16* gp = Ag + (long)(m0 + row) * p.lda + (k0 + kq);
      unsigned la = (unsigned)(size_t)&ldsA[buf][row * LDSP + kq];
      asm volatile("global_load_async_to_lds_b128 %0, %1, off"
                   :: "v"(la), "v"(gp) : "memory");
    }
#pragma unroll
    for (int it = 0; it < BV; ++it) {
      int s = tid + 256 * it;
      int row = s >> 3;              // n
      int kq = (s & 7) << 3;
      const __bf16* gp = Bg + (long)(n0 + row) * p.ldb + (k0 + kq);
      unsigned la = (unsigned)(size_t)&ldsB[buf][row * LDSP + kq];
      asm volatile("global_load_async_to_lds_b128 %0, %1, off"
                   :: "v"(la), "v"(gp) : "memory");
    }
  };

  auto computeStage = [&](int buf) {
#pragma unroll
    for (int ks = 0; ks < KT; ks += 32) {
      v16bf afrag[TI];
#pragma unroll
      for (int ti = 0; ti < TI; ++ti) {
        int m = wr * WM + ti * 16 + l16;
        v8bf lo = *(const v8bf*)&ldsA[buf][m * LDSP + ks + kh * 8];      // K=ks+kh*8+e
        v8bf hi = *(const v8bf*)&ldsA[buf][m * LDSP + ks + 16 + kh * 8]; // +16
        afrag[ti] = CAT16(lo, hi);
      }
      v16bf bfrag[TJ];
#pragma unroll
      for (int tj = 0; tj < TJ; ++tj) {
        int n = wc * WN + tj * 16 + l16;
        v8bf lo = *(const v8bf*)&ldsB[buf][n * LDSP + ks + kh * 8];      // K=ks+kh*8+e
        v8bf hi = *(const v8bf*)&ldsB[buf][n * LDSP + ks + 16 + kh * 8]; // +16
        bfrag[tj] = CAT16(lo, hi);
      }
#pragma unroll
      for (int ti = 0; ti < TI; ++ti)
#pragma unroll
        for (int tj = 0; tj < TJ; ++tj)
          acc[ti][tj] = __builtin_amdgcn_wmma_f32_16x16x32_bf16(
              false, afrag[ti], false, bfrag[tj], (short)0, acc[ti][tj], false, false);
    }
  };

  // ---- software-pipelined main loop (double-buffered LDS, async DMA) ----
  const int nsteps = p.K / KT;
  stage(0, 0);
  int cur = 0;
  for (int step = 0; step < nsteps; ++step) {
    asm volatile("s_wait_asynccnt 0x0" ::: "memory");   // buf[cur] DMA done
    __syncthreads();                                    // publish to workgroup
    if (step + 1 < nsteps) stage(cur ^ 1, (step + 1) * KT);  // DMA behind WMMA
    computeStage(cur);
    cur ^= 1;
  }

  // ---- epilogue: C/D element e -> (m = kh*8 + e, n = l16) ----
#pragma unroll
  for (int ti = 0; ti < TI; ++ti) {
#pragma unroll
    for (int tj = 0; tj < TJ; ++tj) {
      int colg = n0 + wc * WN + tj * 16 + l16;
      float badd = p.bias ? p.bias[colg] : 0.0f;
#pragma unroll
      for (int e = 0; e < 8; ++e) {
        int rowg = m0 + wr * WM + ti * 16 + kh * 8 + e;
        float val = acc[ti][tj][e] * p.alpha + badd;
        if (p.storeCT) {
          ((__bf16*)p.C)[offC + (long)colg * p.ldc + rowg] = (__bf16)val;
        } else if (p.outBf16) {
          ((__bf16*)p.C)[offC + (long)rowg * p.ldc + colg] = (__bf16)val;
        } else {
          ((float*)p.C)[offC + (long)rowg * p.ldc + colg] = val;
        }
      }
    }
  }
}

// ------------------------- softmax with rel_shift --------------------------
// One block per logits row (g = b*H + h in blockIdx.y, i in blockIdx.x).
// rel_shift: out[i,j] = rel[i, T-1-i+j] (j<=i) ; 0 (j==i+1) ; rel[i+1, j-i-2].
// Softmax weights written as bf16 IN PLACE over the f32 content row.

__global__ __launch_bounds__(256) void softmax_shift_kernel(float* __restrict__ logits,
                                                            const __bf16* __restrict__ relb) {
  const int T = Tt;
  const int i = blockIdx.x;
  const int g = blockIdx.y;
  float* crow = logits + ((long)g * T + i) * T;
  const __bf16* r0 = relb + ((long)g * T + i) * T;
  const __bf16* r1 = r0 + T;        // only dereferenced when j > i+1 (=> i+1 < T)
  const int tid = threadIdx.x;

  float vals[4];
  float m = -1e30f;
#pragma unroll
  for (int rix = 0; rix < 4; ++rix) {
    int j = tid + rix * 256;
    float rv;
    if (j <= i)            rv = (float)r0[T - 1 - i + j];
    else if (j == i + 1)   rv = 0.0f;
    else                   rv = (float)r1[j - i - 2];
    float v = crow[j] + rv;
    vals[rix] = v;
    m = fmaxf(m, v);
  }

  __shared__ float red[256];
  red[tid] = m;
  __syncthreads();
  for (int s = 128; s > 0; s >>= 1) {
    if (tid < s) red[tid] = fmaxf(red[tid], red[tid + s]);
    __syncthreads();
  }
  m = red[0];
  __syncthreads();

  float sum = 0.0f;
#pragma unroll
  for (int rix = 0; rix < 4; ++rix) {
    vals[rix] = __expf(vals[rix] - m);
    sum += vals[rix];
  }
  red[tid] = sum;
  __syncthreads();
  for (int s = 128; s > 0; s >>= 1) {
    if (tid < s) red[tid] += red[tid + s];
    __syncthreads();
  }
  float inv = 1.0f / red[0];
  __syncthreads();

  __bf16* w = (__bf16*)crow;        // bf16 weights overlay first half of f32 row
#pragma unroll
  for (int rix = 0; rix < 4; ++rix) {
    int j = tid + rix * 256;
    w[j] = (__bf16)(vals[rix] * inv);
  }
}

// ------------------------------- launcher ----------------------------------

extern "C" void kernel_launch(void* const* d_in, const int* in_sizes, int n_in,
                              void* d_out, int out_size, void* d_ws, size_t ws_size,
                              hipStream_t stream) {
  (void)in_sizes; (void)n_in; (void)out_size; (void)ws_size;

  const float* x    = (const float*)d_in[0];
  const float* pos  = (const float*)d_in[1];
  const float* Wq   = (const float*)d_in[2];
  const float* Wk   = (const float*)d_in[3];
  const float* Wv   = (const float*)d_in[4];
  const float* Wr   = (const float*)d_in[5];
  const float* rwb  = (const float*)d_in[6];   // [H*D]
  const float* rrb  = (const float*)d_in[7];   // [H*D]
  const float* mW   = (const float*)d_in[8];
  const float* mb   = (const float*)d_in[9];
  float* out = (float*)d_out;

  char* ws = (char*)d_ws;
  size_t o = 0;
  auto alloc = [&](size_t bytes) { size_t r = o; o = (o + bytes + 255) & ~(size_t)255; return r; };

  const long NE = BT * Ee;          // 4096*1024
  __bf16* xb   = (__bf16*)(ws + alloc(NE * 2));
  __bf16* posb = (__bf16*)(ws + alloc((long)Tt * Ee * 2));
  __bf16* WqT  = (__bf16*)(ws + alloc((long)Ee * Ee * 2));   // transposed bf16
  __bf16* WkT  = (__bf16*)(ws + alloc((long)Ee * Ee * 2));
  __bf16* WvT  = (__bf16*)(ws + alloc((long)Ee * Ee * 2));
  __bf16* WrT  = (__bf16*)(ws + alloc((long)Ee * Ee * 2));
  __bf16* WmT  = (__bf16*)(ws + alloc((long)Ee * Ee * 2));
  float*  qf   = (float*) (ws + alloc(NE * 4));
  __bf16* qwb  = (__bf16*)(ws + alloc(NE * 2));
  __bf16* qrb  = (__bf16*)(ws + alloc(NE * 2));
  __bf16* kb   = (__bf16*)(ws + alloc(NE * 2));              // [B,T,H,D]
  __bf16* vT   = (__bf16*)(ws + alloc(NE * 2));              // [H*D][B*T]
  __bf16* rb   = (__bf16*)(ws + alloc((long)Tt * Ee * 2));   // [P,H,D]
  __bf16* ab   = (__bf16*)(ws + alloc(NE * 2));              // [B,T,H,D]
  float*  cont = (float*) (ws + alloc(BH * (long)Tt * Tt * 4));   // logits / weights
  __bf16* relb = (__bf16*)(ws + alloc(BH * (long)Tt * Tt * 2));

  // 1) converts: activations plain, weights transposed
  cvt_f32_bf16<<<1024, 256, 0, stream>>>(x,   xb,   NE / 4);
  cvt_f32_bf16<<<512,  256, 0, stream>>>(pos, posb, (long)Tt * Ee / 4);
  dim3 tgrid(Ee / 32, Ee / 32), tblk(32, 8);
  cvt_t_f32_bf16<<<tgrid, tblk, 0, stream>>>(Wq, WqT, Ee, Ee);
  cvt_t_f32_bf16<<<tgrid, tblk, 0, stream>>>(Wk, WkT, Ee, Ee);
  cvt_t_f32_bf16<<<tgrid, tblk, 0, stream>>>(Wv, WvT, Ee, Ee);
  cvt_t_f32_bf16<<<tgrid, tblk, 0, stream>>>(Wr, WrT, Ee, Ee);
  cvt_t_f32_bf16<<<tgrid, tblk, 0, stream>>>(mW, WmT, Ee, Ee);

  const float scale = 0.125f;       // D^-0.5, D = 64
  GemmArgs a;

  // 2) projections (all NT): q (f32, scaled)
  a = { xb, WqT, qf, nullptr, Ee, Ee, Ee, 0,0,0,0,0,0, 1, Ee, scale, 0, 0 };
  gemm_bf16_nt_kernel<128,128,2,4><<<dim3(8,32,1), 256, 0, stream>>>(a);
  // k -> bf16 [B,T,H,D]
  a = { xb, WkT, kb, nullptr, Ee, Ee, Ee, 0,0,0,0,0,0, 1, Ee, 1.0f, 1, 0 };
  gemm_bf16_nt_kernel<128,128,2,4><<<dim3(8,32,1), 256, 0, stream>>>(a);
  // v -> bf16 C-transposed vT[H*D][B*T]
  a = { xb, WvT, vT, nullptr, Ee, Ee, BT, 0,0,0,0,0,0, 1, Ee, 1.0f, 1, 1 };
  gemm_bf16_nt_kernel<128,128,2,4><<<dim3(8,32,1), 256, 0, stream>>>(a);
  // r -> bf16 [P,H,D]
  a = { posb, WrT, rb, nullptr, Ee, Ee, Ee, 0,0,0,0,0,0, 1, Ee, 1.0f, 1, 0 };
  gemm_bf16_nt_kernel<128,128,2,4><<<dim3(8,8,1), 256, 0, stream>>>(a);

  // 3) q + biases -> bf16
  qbias_kernel<<<1024, 256, 0, stream>>>(qf, rwb, rrb, qwb, qrb, NE / 4);

  // 4) content logits: batched NT, f32 out. batch g = b*H + h (inner = H)
  const long sTE = (long)Tt * Ee;
  const long sTT = (long)Tt * Tt;
  a = { qwb, kb, cont, nullptr, Ee, Ee, Tt,
        sTE, Dd, sTE, Dd, (long)Hh * sTT, sTT, Hh, Dd, 1.0f, 0, 0 };
  gemm_bf16_nt_kernel<128,128,2,4><<<dim3(8,8,64), 256, 0, stream>>>(a);

  // 5) relative logits: batched NT, bf16 out (B strides: per-b 0, per-h D)
  a = { qrb, rb, relb, nullptr, Ee, Ee, Tt,
        sTE, Dd, 0, Dd, (long)Hh * sTT, sTT, Hh, Dd, 1.0f, 1, 0 };
  gemm_bf16_nt_kernel<128,128,2,4><<<dim3(8,8,64), 256, 0, stream>>>(a);

  // 6) softmax(content + rel_shift(relative)); bf16 weights in place
  softmax_shift_kernel<<<dim3(Tt, (int)BH), 256, 0, stream>>>(cont, relb);

  // 7) attended = weights . v (NT via vT) -> bf16 [B,T,H,D].
  //    weights live at cont reinterpreted bf16: row stride 2*T elements.
  a = { (const __bf16*)cont, vT, ab, nullptr, 2L * Tt, BT, Ee,
        (long)Hh * sTT * 2, sTT * 2, (long)Tt, (long)Dd * BT, sTE, Dd,
        Hh, Tt, 1.0f, 1, 0 };
  gemm_bf16_nt_kernel<128,64,4,2><<<dim3(1,8,64), 256, 0, stream>>>(a);

  // 8) merge: out = attended @ merge_W + merge_b (NT via WmT, f32)
  a = { ab, WmT, out, mb, Ee, Ee, Ee, 0,0,0,0,0,0, 1, Ee, 1.0f, 0, 0 };
  gemm_bf16_nt_kernel<128,128,2,4><<<dim3(8,32,1), 256, 0, stream>>>(a);
}
